// SAGE_644245095091
// MI455X (gfx1250) — compile-verified, matrix operand order checked by reading, not run on hardware
//
#include <hip/hip_runtime.h>

typedef __attribute__((ext_vector_type(16))) _Float16 v16h;
typedef __attribute__((ext_vector_type(8)))  float    v8f;

#define NODES 100000
#define EDGES 1600000
#define FIN   128
#define FHID  128
#define FOUT  64
#define BN_EPS 1e-5f

// ---------------------------------------------------------------------------
// Degree: deg[dst] += 1
// ---------------------------------------------------------------------------
__global__ void deg_kernel(const int* __restrict__ dst, float* __restrict__ deg, int E) {
    int e = blockIdx.x * blockDim.x + threadIdx.x;
    if (e < E) atomicAdd(&deg[dst[e]], 1.0f);
}

// deg[i] = 1 / max(deg[i], 1)
__global__ void invdeg_kernel(float* __restrict__ deg, int n) {
    int i = blockIdx.x * blockDim.x + threadIdx.x;
    if (i < n) deg[i] = 1.0f / fmaxf(deg[i], 1.0f);
}

// ---------------------------------------------------------------------------
// Edge scatter: agg[dst] += x[src]   (128 floats per edge, 32 lanes x float4)
// ---------------------------------------------------------------------------
__global__ __launch_bounds__(256) void scatter_add_kernel(
    const float* __restrict__ x, const int* __restrict__ src,
    const int* __restrict__ dst, float* __restrict__ agg, int E)
{
    int tid  = blockIdx.x * blockDim.x + threadIdx.x;
    int e    = tid >> 5;
    int lane = tid & 31;
    if (e >= E) return;
    size_t sb = (size_t)src[e] * FIN + lane * 4;
    size_t db = (size_t)dst[e] * FIN + lane * 4;
    float4 v = *reinterpret_cast<const float4*>(x + sb);
    atomicAdd(agg + db + 0, v.x);
    atomicAdd(agg + db + 1, v.y);
    atomicAdd(agg + db + 2, v.z);
    atomicAdd(agg + db + 3, v.w);
}

// ---------------------------------------------------------------------------
// Weight pre-pack: stacked [Wself;Wneigh] (256 x H, row-major f32) ->
// WMMA B-operand fragment layout, f16:
//   Wpack[((ct*8 + kt)*32 + lane)*16 + e]
// where lane = hi*16 + m holds column (ct*16+m), element e = half*8 + j maps
// to K = kt*32 + hi*8 + half*16 + j   (ISA 16-bit B 32x16 layout).
// Each consuming lane then reads its 16 halves as one contiguous 32B chunk.
// ---------------------------------------------------------------------------
__global__ void pack_weights_kernel(const float* __restrict__ Wself,
                                    const float* __restrict__ Wneigh,
                                    _Float16* __restrict__ Wpack, int H)
{
    int idx = blockIdx.x * blockDim.x + threadIdx.x;   // total = 256*H
    if (idx >= 256 * H) return;
    int e    = idx & 15;
    int lane = (idx >> 4) & 31;
    int kt   = (idx >> 9) & 7;
    int ct   = idx >> 12;
    int m    = lane & 15;
    int hi   = lane >> 4;
    int j    = e & 7;
    int half = e >> 3;
    int K    = kt * 32 + hi * 8 + half * 16 + j;
    int col  = ct * 16 + m;
    float v  = (K < FIN) ? Wself[(size_t)K * H + col]
                         : Wneigh[(size_t)(K - FIN) * H + col];
    Wpack[idx] = (_Float16)v;
}

// ---------------------------------------------------------------------------
// Fused SAGE GEMM:  out[N,H] = [x | agg*invdeg] (N,256) @ Wpack(256,H) + b
// One wave -> one 16x16 tile via v_wmma_f32_16x16x32_f16 (K=256 -> 8 WMMAs).
// 10 waves/block, 625 blocks -> exactly 6250 row tiles (100000 rows).
// A fragments in registers across all column tiles (float4 global loads);
// B fragments are single contiguous 32B per-lane global loads (L2-resident).
// No LDS, no barriers.
// ---------------------------------------------------------------------------
#define CVT8(a, base, p0, p1, s)                                   \
    do {                                                           \
        a[(base) + 0] = (_Float16)((p0).x * (s));                  \
        a[(base) + 1] = (_Float16)((p0).y * (s));                  \
        a[(base) + 2] = (_Float16)((p0).z * (s));                  \
        a[(base) + 3] = (_Float16)((p0).w * (s));                  \
        a[(base) + 4] = (_Float16)((p1).x * (s));                  \
        a[(base) + 5] = (_Float16)((p1).y * (s));                  \
        a[(base) + 6] = (_Float16)((p1).z * (s));                  \
        a[(base) + 7] = (_Float16)((p1).w * (s));                  \
    } while (0)

__global__ __launch_bounds__(320) void sage_gemm_wmma(
    const float* __restrict__ xs,        // [N,128] self features
    const float* __restrict__ agg,       // [N,128] neighbor sums
    const float* __restrict__ invdeg,    // [N]
    const _Float16* __restrict__ Wpack,  // [H/16][8][32][16] fragment-ready f16
    const float* __restrict__ bias,      // [H]
    float* __restrict__ out,             // [N,H]
    int H)
{
    const int lane = threadIdx.x & 31;
    const int wave = threadIdx.x >> 5;           // 0..9
    const int m    = lane & 15;
    const int hi   = lane >> 4;                  // ISA K-split select
    const int row_tile = blockIdx.x * 10 + wave; // 0..6249
    const int r = row_tile * 16 + m;             // this lane's A row

    const float idr = invdeg[r];
    const float4* xr4 = (const float4*)(xs  + (size_t)r * FIN);
    const float4* ar4 = (const float4*)(agg + (size_t)r * FIN);

    // A fragments: lane m holds row m, K = kb..kb+7 and kb+16..kb+23 (kb = kt*32+hi*8)
    v16h afrag[8];
#pragma unroll
    for (int kt = 0; kt < 8; ++kt) {
        const int kb = kt * 32 + hi * 8;         // multiple of 8
        v16h a;
        if (kt < 4) {                            // K < 128 -> self features
            const float4 p0 = xr4[kb >> 2];
            const float4 p1 = xr4[(kb >> 2) + 1];
            const float4 q0 = xr4[(kb + 16) >> 2];
            const float4 q1 = xr4[((kb + 16) >> 2) + 1];
            CVT8(a, 0, p0, p1, 1.0f);
            CVT8(a, 8, q0, q1, 1.0f);
        } else {                                 // K >= 128 -> mean neighbor
            const int ab = kb - FIN;
            const float4 p0 = ar4[ab >> 2];
            const float4 p1 = ar4[(ab >> 2) + 1];
            const float4 q0 = ar4[(ab + 16) >> 2];
            const float4 q1 = ar4[((ab + 16) >> 2) + 1];
            CVT8(a, 0, p0, p1, idr);
            CVT8(a, 8, q0, q1, idr);
        }
        afrag[kt] = a;
    }

    const v16h* bfrag = (const v16h*)Wpack;      // [ct*256 + kt*32 + lane]
    const int ntiles = H >> 4;
    for (int ct = 0; ct < ntiles; ++ct) {
        const int col = ct * 16 + m;
        v8f acc = {};
#pragma unroll
        for (int kt = 0; kt < 8; ++kt) {
            v16h b = bfrag[(size_t)ct * 256 + kt * 32 + lane];
            acc = __builtin_amdgcn_wmma_f32_16x16x32_f16(
                false, afrag[kt], false, b, (short)0, acc, false, false);
        }
        // C/D layout: lanes 0-15: VGPR v -> M=v; lanes 16-31: M=v+8; N = lane%16
        const float bv = bias[col];
#pragma unroll
        for (int v = 0; v < 8; ++v) {
            int rr = row_tile * 16 + hi * 8 + v;
            out[(size_t)rr * H + col] = acc[v] + bv;
        }
    }
}

// ---------------------------------------------------------------------------
// Column stats: stats[c] += sum, stats[H+c] += sum of squares (256 rows/block)
// ---------------------------------------------------------------------------
__global__ void col_stats_kernel(const float* __restrict__ x, float* __restrict__ stats,
                                 int H, int nrows)
{
    int c = threadIdx.x;                       // blockDim.x == H
    long r0 = (long)blockIdx.x * 256;
    long r1 = r0 + 256; if (r1 > nrows) r1 = nrows;
    float s = 0.f, s2 = 0.f;
    for (long r = r0; r < r1; ++r) {
        float v = x[r * H + c];
        s += v; s2 += v * v;
    }
    atomicAdd(&stats[c], s);
    atomicAdd(&stats[H + c], s2);
}

// coeff[c] = gamma*rsqrt(var+eps), coeff[H+c] = beta - mu*coeff[c]
__global__ void bn_coeff_kernel(const float* __restrict__ stats,
                                const float* __restrict__ gamma,
                                const float* __restrict__ beta,
                                float* __restrict__ coeff, int H, float invN)
{
    int c = threadIdx.x;
    if (c >= H) return;
    float mu  = stats[c] * invN;
    float var = stats[H + c] * invN - mu * mu;
    float sc  = gamma[c] * rsqrtf(var + BN_EPS);
    coeff[c]     = sc;
    coeff[H + c] = beta[c] - mu * sc;
}

// x = relu(x*scale + shift), in place
__global__ void bn_relu_kernel(float* __restrict__ x, const float* __restrict__ coeff,
                               int H, long total)
{
    long i = (long)blockIdx.x * blockDim.x + threadIdx.x;
    if (i >= total) return;
    int c = (int)(i & (long)(H - 1));          // H is a power of two (128)
    float v = x[i] * coeff[c] + coeff[H + c];
    x[i] = fmaxf(v, 0.f);
}

// ---------------------------------------------------------------------------
// Row-wise log_softmax over 64 columns: one wave per row, 2 elements per lane
// ---------------------------------------------------------------------------
__global__ __launch_bounds__(256) void log_softmax64_kernel(
    const float* __restrict__ x, float* __restrict__ out, int nrows)
{
    int tid  = blockIdx.x * blockDim.x + threadIdx.x;
    int row  = tid >> 5;
    int lane = tid & 31;
    if (row >= nrows) return;
    const float* xr = x + (size_t)row * FOUT;
    float v0 = xr[lane];
    float v1 = xr[lane + 32];
    float mx = fmaxf(v0, v1);
#pragma unroll
    for (int o = 16; o > 0; o >>= 1) mx = fmaxf(mx, __shfl_xor(mx, o, 32));
    float s = __expf(v0 - mx) + __expf(v1 - mx);
#pragma unroll
    for (int o = 16; o > 0; o >>= 1) s += __shfl_xor(s, o, 32);
    float l = mx + __logf(s);
    out[(size_t)row * FOUT + lane]      = v0 - l;
    out[(size_t)row * FOUT + lane + 32] = v1 - l;
}

// ---------------------------------------------------------------------------
extern "C" void kernel_launch(void* const* d_in, const int* in_sizes, int n_in,
                              void* d_out, int out_size, void* d_ws, size_t ws_size,
                              hipStream_t stream)
{
    (void)in_sizes; (void)n_in; (void)out_size; (void)ws_size;

    // Inputs per setup_inputs() order
    const float* inputs   = (const float*)d_in[0];   // [N,128]
    const int*   src      = (const int*)  d_in[1];   // [E]
    const int*   dst      = (const int*)  d_in[2];   // [E]
    const float* w_self1  = (const float*)d_in[3];
    const float* w_neigh1 = (const float*)d_in[4];
    const float* b1       = (const float*)d_in[5];
    const float* w_self2  = (const float*)d_in[6];
    const float* w_neigh2 = (const float*)d_in[7];
    const float* b2       = (const float*)d_in[8];
    const float* w_self3  = (const float*)d_in[9];
    const float* w_neigh3 = (const float*)d_in[10];
    const float* b3       = (const float*)d_in[11];
    const float* gamma1   = (const float*)d_in[12];
    const float* beta1    = (const float*)d_in[13];
    const float* gamma2   = (const float*)d_in[14];
    const float* beta2    = (const float*)d_in[15];

    // Output regions: (log_softmax [N,64], first_l_x [N,128], feat [N,128])
    float* out_ls    = (float*)d_out;
    float* out_first = out_ls + (size_t)NODES * FOUT;
    float* out_feat  = out_first + (size_t)NODES * FIN;

    // Workspace layout (all chunks 32B-aligned: every size is a multiple of 8 floats)
    float* ws     = (float*)d_ws;
    float* invdeg = ws;                                   // N
    float* aggbuf = invdeg + NODES;                       // N*128
    float* x1     = aggbuf + (size_t)NODES * FIN;         // N*128
    float* x3     = x1 + (size_t)NODES * FHID;            // N*64
    float* stats  = x3 + (size_t)NODES * FOUT;            // 2*128
    float* coeff  = stats + 2 * FHID;                     // 2*128
    _Float16* pack1 = (_Float16*)(coeff + 2 * FHID);      // 256*128 halves (64KB)
    _Float16* pack2 = pack1 + 256 * FHID;                 // 256*128 halves
    _Float16* pack3 = pack2 + 256 * FHID;                 // 256*64  halves

    const int thr = 256;
    const int scatterBlocks = (EDGES * 32) / thr;         // 200000
    const int gemmBlocks    = NODES / (10 * 16);          // 625
    const int statBlocks    = (NODES + 255) / 256;        // 391
    const int bnBlocks      = (NODES * FHID) / thr;       // 50000
    const int lsBlocks      = (NODES * 32) / thr;         // 12500

    // ---- degrees + packed weights (once per launch) ----
    hipMemsetAsync(invdeg, 0, NODES * sizeof(float), stream);
    deg_kernel<<<(EDGES + thr - 1) / thr, thr, 0, stream>>>(dst, invdeg, EDGES);
    invdeg_kernel<<<(NODES + thr - 1) / thr, thr, 0, stream>>>(invdeg, NODES);
    pack_weights_kernel<<<FHID, thr, 0, stream>>>(w_self1, w_neigh1, pack1, FHID);
    pack_weights_kernel<<<FHID, thr, 0, stream>>>(w_self2, w_neigh2, pack2, FHID);
    pack_weights_kernel<<<FOUT, thr, 0, stream>>>(w_self3, w_neigh3, pack3, FOUT);

    // ---- first_l_x output = inputs ----
    hipMemcpyAsync(out_first, inputs, (size_t)NODES * FIN * sizeof(float),
                   hipMemcpyDeviceToDevice, stream);

    // ================= Layer 1 =================
    hipMemsetAsync(aggbuf, 0, (size_t)NODES * FIN * sizeof(float), stream);
    scatter_add_kernel<<<scatterBlocks, thr, 0, stream>>>(inputs, src, dst, aggbuf, EDGES);
    sage_gemm_wmma<<<gemmBlocks, 320, 0, stream>>>(inputs, aggbuf, invdeg,
                                                   pack1, b1, x1, FHID);
    hipMemsetAsync(stats, 0, 2 * FHID * sizeof(float), stream);
    col_stats_kernel<<<statBlocks, FHID, 0, stream>>>(x1, stats, FHID, NODES);
    bn_coeff_kernel<<<1, FHID, 0, stream>>>(stats, gamma1, beta1, coeff, FHID, 1.0f / NODES);
    bn_relu_kernel<<<bnBlocks, thr, 0, stream>>>(x1, coeff, FHID, (long)NODES * FHID);

    // ================= Layer 2 =================
    hipMemsetAsync(aggbuf, 0, (size_t)NODES * FHID * sizeof(float), stream);
    scatter_add_kernel<<<scatterBlocks, thr, 0, stream>>>(x1, src, dst, aggbuf, EDGES);
    sage_gemm_wmma<<<gemmBlocks, 320, 0, stream>>>(x1, aggbuf, invdeg,
                                                   pack2, b2, out_feat, FHID);
    hipMemsetAsync(stats, 0, 2 * FHID * sizeof(float), stream);
    col_stats_kernel<<<statBlocks, FHID, 0, stream>>>(out_feat, stats, FHID, NODES);
    bn_coeff_kernel<<<1, FHID, 0, stream>>>(stats, gamma2, beta2, coeff, FHID, 1.0f / NODES);
    bn_relu_kernel<<<bnBlocks, thr, 0, stream>>>(out_feat, coeff, FHID, (long)NODES * FHID);

    // ================= Layer 3 + log_softmax =================
    hipMemsetAsync(aggbuf, 0, (size_t)NODES * FHID * sizeof(float), stream);
    scatter_add_kernel<<<scatterBlocks, thr, 0, stream>>>(out_feat, src, dst, aggbuf, EDGES);
    sage_gemm_wmma<<<gemmBlocks, 320, 0, stream>>>(out_feat, aggbuf, invdeg,
                                                   pack3, b3, x3, FOUT);
    log_softmax64_kernel<<<lsBlocks, thr, 0, stream>>>(x3, out_ls, NODES);
}
